// DotProductAttention_56358560858799
// MI455X (gfx1250) — matrix operand order repaired and checked by reading, not compiled, not run on hardware
//
#include <hip/hip_runtime.h>

// ---------- WMMA vector types (gfx1250, wave32) ----------
typedef __attribute__((ext_vector_type(16))) __bf16       v16bf;
typedef __attribute__((ext_vector_type(8)))  float        v8f;
typedef __attribute__((ext_vector_type(4)))  unsigned int u32x4;

// fp32 -> bf16 round-to-nearest-even
__device__ __forceinline__ unsigned short f2bf(float f) {
  union { float f; unsigned u; } c; c.f = f;
  unsigned u = c.u;
  u += 0x7FFFu + ((u >> 16) & 1u);
  return (unsigned short)(u >> 16);
}

union BOp { v16bf v; u32x4 q[2]; unsigned short s[16]; };

#define D_DIM 64   // head dim
#define KT_F  64   // keys / iteration, fast kernel
#define KT_S  32   // keys / iteration, staged fallback

// =====================================================================
// Pre-conversion: Kb[b][k][d] = bf16(K), Vt[b][d][k] = bf16(V) (transposed)
// Both layouts feed WMMA B-operands with 32B contiguous loads.
// =====================================================================
__global__ __launch_bounds__(128) void convert_kv(
    const float* __restrict__ Kg, const float* __restrict__ Vg,
    unsigned short* __restrict__ Kb, unsigned short* __restrict__ Vt, int Lk)
{
  __shared__ __attribute__((aligned(32))) unsigned short vt[D_DIM * 32];
  const int b  = blockIdx.y;
  const int k0 = blockIdx.x * 32;
  const int t  = threadIdx.x;
  {
    const int key = t >> 2, cb = (t & 3) * 16;
    const float* kp = Kg + ((size_t)b * Lk + k0 + key) * D_DIM + cb;
    const float* vp = Vg + ((size_t)b * Lk + k0 + key) * D_DIM + cb;
    unsigned short* ko = Kb + ((size_t)b * Lk + k0 + key) * D_DIM + cb;
#pragma unroll
    for (int i = 0; i < 16; ++i) ko[i] = f2bf(kp[i]);
#pragma unroll
    for (int i = 0; i < 16; ++i) vt[(cb + i) * 32 + key] = f2bf(vp[i]);
  }
  __syncthreads();
  {
    const int ch = t >> 1, hf = t & 1;
    unsigned short* vo = Vt + ((size_t)b * D_DIM + ch) * Lk + k0 + hf * 16;
#pragma unroll
    for (int i = 0; i < 16; ++i) vo[i] = vt[ch * 32 + hf * 16 + i];
  }
}

// =====================================================================
// Fast flash-attention: B operands straight from global bf16 tables.
// Block = 256 threads = 8 waves; each wave owns 16 Q rows.
// =====================================================================
__global__ __launch_bounds__(256) void fa_fast(
    const float* __restrict__ Qg, const unsigned short* __restrict__ Kb,
    const unsigned short* __restrict__ Vt, const int* __restrict__ vlen,
    float* __restrict__ Og, int Lq, int Lk)
{
  __shared__ __attribute__((aligned(32))) unsigned short pbuf[8 * 16 * KT_F];

  const int b     = blockIdx.y;
  const int qbase = blockIdx.x * 128;
  const int tid   = threadIdx.x;
  const int wave  = tid >> 5;
  const int lane  = tid & 31;
  const int ln    = lane & 15;
  const int hi    = lane >> 4;

  const int vl  = vlen[b];
  const int nkt = (vl + KT_F - 1) / KT_F;

  // Q tile (16 rows / wave) in A-operand layout, pre-scaled by 1/sqrt(D)
  const int    qrow = qbase + wave * 16 + ln;
  const float* qp   = Qg + ((size_t)b * Lq + qrow) * D_DIM;
  BOp qa[2];
#pragma unroll
  for (int c = 0; c < 2; ++c)
#pragma unroll
    for (int e = 0; e < 8; ++e) {
      qa[c].s[e]     = f2bf(qp[32 * c + 16 * hi +     e] * 0.125f);
      qa[c].s[e + 8] = f2bf(qp[32 * c + 16 * hi + 8 + e] * 0.125f);
    }

  v8f   o[4];
  float m_s[8], l_s[8];
#pragma unroll
  for (int g = 0; g < 4; ++g)
#pragma unroll
    for (int r = 0; r < 8; ++r) o[g][r] = 0.0f;
#pragma unroll
  for (int r = 0; r < 8; ++r) { m_s[r] = -1e30f; l_s[r] = 0.0f; }

  const unsigned short* kbase = Kb + (size_t)b * Lk * D_DIM;
  const unsigned short* vbase = Vt + (size_t)b * D_DIM * Lk;
  unsigned short*       pw    = &pbuf[wave * (16 * KT_F)];

  auto tile = [&](int k0, bool masked) {
    // ---- S = Q.K^T : 4 key-groups x (K-dim 64 = 2 WMMA)
    v8f s[4];
#pragma unroll
    for (int g2 = 0; g2 < 4; ++g2) {
      const unsigned short* kbp = kbase + (size_t)(k0 + g2 * 16 + ln) * D_DIM + 16 * hi;
      BOp kb0, kb1;
      kb0.q[0] = *(const u32x4*)(kbp);
      kb0.q[1] = *(const u32x4*)(kbp + 8);
      kb1.q[0] = *(const u32x4*)(kbp + 32);
      kb1.q[1] = *(const u32x4*)(kbp + 40);
      v8f acc;
#pragma unroll
      for (int r = 0; r < 8; ++r) acc[r] = 0.0f;
      acc = __builtin_amdgcn_wmma_f32_16x16x32_bf16(false, qa[0].v, false, kb0.v,
                                                    (short)0, acc, false, false);
      acc = __builtin_amdgcn_wmma_f32_16x16x32_bf16(false, qa[1].v, false, kb1.v,
                                                    (short)0, acc, false, false);
      s[g2] = acc;
    }

    // ---- key-length mask: only the peeled last tile pays for it
    if (masked) {
#pragma unroll
      for (int g2 = 0; g2 < 4; ++g2) {
        const int j = k0 + g2 * 16 + ln;
#pragma unroll
        for (int r = 0; r < 8; ++r)
          if (j >= vl) s[g2][r] = -1e30f;
      }
    }

    // ---- online softmax, fully unrolled (no dynamic VGPR indexing!)
#pragma unroll
    for (int r = 0; r < 8; ++r) {
      float mx = fmaxf(fmaxf(s[0][r], s[1][r]), fmaxf(s[2][r], s[3][r]));
      mx = fmaxf(mx, __shfl_xor(mx, 1));
      mx = fmaxf(mx, __shfl_xor(mx, 2));
      mx = fmaxf(mx, __shfl_xor(mx, 4));
      mx = fmaxf(mx, __shfl_xor(mx, 8));
      const float mn   = fmaxf(m_s[r], mx);
      const float corr = __expf(m_s[r] - mn);
      float p[4], sum = 0.0f;
#pragma unroll
      for (int g2 = 0; g2 < 4; ++g2) { p[g2] = __expf(s[g2][r] - mn); sum += p[g2]; }
      sum += __shfl_xor(sum, 1);
      sum += __shfl_xor(sum, 2);
      sum += __shfl_xor(sum, 4);
      sum += __shfl_xor(sum, 8);
      l_s[r] = l_s[r] * corr + sum;
      m_s[r] = mn;
#pragma unroll
      for (int g = 0; g < 4; ++g) o[g][r] *= corr;
      const int row = r + 8 * hi;
#pragma unroll
      for (int g2 = 0; g2 < 4; ++g2) pw[row * KT_F + g2 * 16 + ln] = f2bf(p[g2]);
    }

    // ---- P back in A-operand layout (wave-private LDS, in-order)
    BOp pA[2];
#pragma unroll
    for (int c = 0; c < 2; ++c) {
      const unsigned short* pr = &pw[ln * KT_F + 32 * c + 16 * hi];
      pA[c].q[0] = *(const u32x4*)(pr);
      pA[c].q[1] = *(const u32x4*)(pr + 8);
    }

    // ---- O += P.V : 4 channel-groups x 2 key-chunks
#pragma unroll
    for (int g = 0; g < 4; ++g) {
      const unsigned short* vbp = vbase + (size_t)(g * 16 + ln) * Lk + k0 + 16 * hi;
      BOp vb0, vb1;
      vb0.q[0] = *(const u32x4*)(vbp);
      vb0.q[1] = *(const u32x4*)(vbp + 8);
      vb1.q[0] = *(const u32x4*)(vbp + 32);
      vb1.q[1] = *(const u32x4*)(vbp + 40);
      o[g] = __builtin_amdgcn_wmma_f32_16x16x32_bf16(false, pA[0].v, false, vb0.v,
                                                     (short)0, o[g], false, false);
      o[g] = __builtin_amdgcn_wmma_f32_16x16x32_bf16(false, pA[1].v, false, vb1.v,
                                                     (short)0, o[g], false, false);
    }
  };

  for (int kt = 0; kt < nkt - 1; ++kt) {
    const int nk0 = (kt + 1) * KT_F;
    // prefetch next tile (speculative, lane-spread across rows)
    __builtin_prefetch(kbase + (size_t)(nk0 + lane) * D_DIM, 0, 0);
    __builtin_prefetch(kbase + (size_t)(nk0 + 32 + lane) * D_DIM, 0, 0);
    __builtin_prefetch(vbase + (size_t)(lane * 2) * Lk + nk0, 0, 0);
    __builtin_prefetch(vbase + (size_t)(lane * 2 + 1) * Lk + nk0, 0, 0);
    tile(kt * KT_F, false);
  }
  tile((nkt - 1) * KT_F, true);   // only tile that pays for masking

  // ---- finalize
#pragma unroll
  for (int r = 0; r < 8; ++r) {
    const float inv = 1.0f / l_s[r];
    const int   row = qbase + wave * 16 + r + 8 * hi;
    float* op = Og + ((size_t)b * Lq + row) * D_DIM;
#pragma unroll
    for (int g = 0; g < 4; ++g)
      op[g * 16 + ln] = o[g][r] * inv;
  }
}

// =====================================================================
// Fallback (ws too small): LDS-staged version, fully unrolled
// =====================================================================
__global__ __launch_bounds__(128) void fa_staged(
    const float* __restrict__ Qg, const float* __restrict__ Kg,
    const float* __restrict__ Vg, const int* __restrict__ vlen,
    float* __restrict__ Og, int Lq, int Lk)
{
  __shared__ __attribute__((aligned(32))) unsigned short kbuf[KT_S * D_DIM];
  __shared__ __attribute__((aligned(32))) unsigned short vtbuf[D_DIM * KT_S];
  __shared__ __attribute__((aligned(32))) unsigned short pbuf[4 * 16 * KT_S];

  const int b = blockIdx.y, qbase = blockIdx.x * 64;
  const int tid = threadIdx.x, wave = tid >> 5, lane = tid & 31;
  const int ln = lane & 15, hi = lane >> 4;
  const int vl = vlen[b];
  const int nkt = (vl + KT_S - 1) / KT_S;

  const int qrow = qbase + wave * 16 + ln;
  const float* qp = Qg + ((size_t)b * Lq + qrow) * D_DIM;
  BOp qa[2];
#pragma unroll
  for (int c = 0; c < 2; ++c)
#pragma unroll
    for (int e = 0; e < 8; ++e) {
      qa[c].s[e]     = f2bf(qp[32 * c + 16 * hi +     e] * 0.125f);
      qa[c].s[e + 8] = f2bf(qp[32 * c + 16 * hi + 8 + e] * 0.125f);
    }

  v8f o[4]; float m_s[8], l_s[8];
#pragma unroll
  for (int g = 0; g < 4; ++g)
#pragma unroll
    for (int r = 0; r < 8; ++r) o[g][r] = 0.0f;
#pragma unroll
  for (int r = 0; r < 8; ++r) { m_s[r] = -1e30f; l_s[r] = 0.0f; }

  for (int kt = 0; kt < nkt; ++kt) {
    const int k0 = kt * KT_S;
    {
      const int key = tid >> 2, cb = (tid & 3) * 16;
      const float* kp = Kg + ((size_t)b * Lk + k0 + key) * D_DIM + cb;
      const float* vp = Vg + ((size_t)b * Lk + k0 + key) * D_DIM + cb;
#pragma unroll
      for (int i = 0; i < 16; ++i) {
        kbuf[key * D_DIM + cb + i] = f2bf(kp[i]);
        vtbuf[(cb + i) * KT_S + key] = f2bf(vp[i]);
      }
    }
    __syncthreads();

    v8f s[2];
#pragma unroll
    for (int g2 = 0; g2 < 2; ++g2) {
      const unsigned short* kbp = &kbuf[(g2 * 16 + ln) * D_DIM + 16 * hi];
      BOp kb0, kb1;
      kb0.q[0] = *(const u32x4*)(kbp);
      kb0.q[1] = *(const u32x4*)(kbp + 8);
      kb1.q[0] = *(const u32x4*)(kbp + 32);
      kb1.q[1] = *(const u32x4*)(kbp + 40);
      v8f acc;
#pragma unroll
      for (int r = 0; r < 8; ++r) acc[r] = 0.0f;
      acc = __builtin_amdgcn_wmma_f32_16x16x32_bf16(false, qa[0].v, false, kb0.v, (short)0, acc, false, false);
      acc = __builtin_amdgcn_wmma_f32_16x16x32_bf16(false, qa[1].v, false, kb1.v, (short)0, acc, false, false);
      s[g2] = acc;
    }

    const int j0 = k0 + ln, j1 = k0 + 16 + ln;
#pragma unroll
    for (int r = 0; r < 8; ++r) {
      if (j0 >= vl) s[0][r] = -1e30f;
      if (j1 >= vl) s[1][r] = -1e30f;
    }

    unsigned short* pw = &pbuf[wave * (16 * KT_S)];
#pragma unroll
    for (int r = 0; r < 8; ++r) {
      float mx = fmaxf(s[0][r], s[1][r]);
      mx = fmaxf(mx, __shfl_xor(mx, 1));
      mx = fmaxf(mx, __shfl_xor(mx, 2));
      mx = fmaxf(mx, __shfl_xor(mx, 4));
      mx = fmaxf(mx, __shfl_xor(mx, 8));
      const float mn = fmaxf(m_s[r], mx);
      const float corr = __expf(m_s[r] - mn);
      const float p0 = __expf(s[0][r] - mn);
      const float p1 = __expf(s[1][r] - mn);
      float sum = p0 + p1;
      sum += __shfl_xor(sum, 1);
      sum += __shfl_xor(sum, 2);
      sum += __shfl_xor(sum, 4);
      sum += __shfl_xor(sum, 8);
      l_s[r] = l_s[r] * corr + sum;
      m_s[r] = mn;
#pragma unroll
      for (int g = 0; g < 4; ++g) o[g][r] *= corr;
      const int row = r + 8 * hi;
      pw[row * KT_S + ln]      = f2bf(p0);
      pw[row * KT_S + 16 + ln] = f2bf(p1);
    }

    BOp pA;
    const unsigned short* pr = &pw[ln * KT_S + 16 * hi];
    pA.q[0] = *(const u32x4*)(pr);
    pA.q[1] = *(const u32x4*)(pr + 8);

#pragma unroll
    for (int g = 0; g < 4; ++g) {
      const unsigned short* vbp = &vtbuf[(g * 16 + ln) * KT_S + 16 * hi];
      BOp vb;
      vb.q[0] = *(const u32x4*)(vbp);
      vb.q[1] = *(const u32x4*)(vbp + 8);
      o[g] = __builtin_amdgcn_wmma_f32_16x16x32_bf16(false, pA.v, false, vb.v, (short)0, o[g], false, false);
    }
    __syncthreads();
  }

#pragma unroll
  for (int r = 0; r < 8; ++r) {
    const float inv = 1.0f / l_s[r];
    const int row = qbase + wave * 16 + r + 8 * hi;
    float* op = Og + ((size_t)b * Lq + row) * D_DIM;
#pragma unroll
    for (int g = 0; g < 4; ++g)
      op[g * 16 + ln] = o[g][r] * inv;
  }
}

extern "C" void kernel_launch(void* const* d_in, const int* in_sizes, int n_in,
                              void* d_out, int out_size, void* d_ws, size_t ws_size,
                              hipStream_t stream) {
  const float* Q    = (const float*)d_in[0];
  const float* K    = (const float*)d_in[1];
  const float* V    = (const float*)d_in[2];
  const int*   vlen = (const int*)d_in[3];
  float*       out  = (float*)d_out;

  const int B  = in_sizes[3];
  const int Lq = in_sizes[0] / (B * D_DIM);
  const int Lk = in_sizes[1] / (B * D_DIM);

  const size_t tbl = (size_t)B * Lk * D_DIM * sizeof(unsigned short);
  if (ws_size >= 2 * tbl) {
    unsigned short* Kb = (unsigned short*)d_ws;
    unsigned short* Vt = (unsigned short*)((char*)d_ws + tbl);
    convert_kv<<<dim3(Lk / 32, B), 128, 0, stream>>>(K, V, Kb, Vt, Lk);
    fa_fast<<<dim3(Lq / 128, B), 256, 0, stream>>>(Q, Kb, Vt, vlen, out, Lq, Lk);
  } else {
    fa_staged<<<dim3(Lq / 64, B), 128, 0, stream>>>(Q, K, V, vlen, out, Lq, Lk);
  }
}
